// MultiHeadAttention_38835094290526
// MI455X (gfx1250) — matrix operand order repaired and checked by reading, not compiled, not run on hardware
//
#include <hip/hip_runtime.h>
#include <hip/hip_bf16.h>

// ---------------------------------------------------------------------------
// MHA on gfx1250 (MI455X): bf16 WMMA + async global->LDS staging (ASYNCcnt),
// flash-style attention.  B=2, S=2048, D=1024, H=16, Dh=64
// ---------------------------------------------------------------------------

typedef __attribute__((ext_vector_type(16))) __bf16 v16bf;
typedef __attribute__((ext_vector_type(8)))  __bf16 v8bf;
typedef __attribute__((ext_vector_type(8)))  float  v8f;

#define WMMA_BF16(a, b, c) \
  __builtin_amdgcn_wmma_f32_16x16x32_bf16(false, (a), false, (b), (short)0, (c), false, false)

// async global -> LDS copy of 16 bytes per lane (ASYNCcnt-tracked)
__device__ inline void async_b128(unsigned lds_off, const void* gaddr) {
  asm volatile("global_load_async_to_lds_b128 %0, %1, off"
               :: "v"(lds_off), "v"(gaddr) : "memory");
}
#define WAIT_ASYNC(n) asm volatile("s_wait_asynccnt %0" :: "i"(n) : "memory")

static constexpr int Bn = 2, S = 2048, D = 1024, H = 16, Dh = 64;
static constexpr int M = Bn * S;  // 4096 rows

// ---- workspace layout (bytes) ---------------------------------------------
static constexpr size_t OFF_XB  = 0;                             // x bf16: 8 MiB
static constexpr size_t OFF_WTQ = OFF_XB  + (size_t)M * D * 2;   // W^T bf16: 2 MiB each
static constexpr size_t OFF_WTK = OFF_WTQ + (size_t)D * D * 2;
static constexpr size_t OFF_WTV = OFF_WTK + (size_t)D * D * 2;
static constexpr size_t OFF_WTO = OFF_WTV + (size_t)D * D * 2;
static constexpr size_t OFF_QB  = OFF_WTO + (size_t)D * D * 2;   // Q [B,H,S,Dh]
static constexpr size_t OFF_KB  = OFF_QB  + (size_t)M * D * 2;   // K [B,H,S,Dh]
static constexpr size_t OFF_VT  = OFF_KB  + (size_t)M * D * 2;   // V^T [B,H,Dh,S]
static constexpr size_t OFF_CTX = OFF_VT  + (size_t)M * D * 2;   // ctx [B,S,D]
static constexpr size_t WS_NEED = OFF_CTX + (size_t)M * D * 2;   // 48 MiB

// ---------------------------------------------------------------------------
// Operand loaders matching ISA 7.12.2 VGPR layouts (wave32).
__device__ inline v16bf load_a16x32(const __bf16* src, int lda) {
  const int lane = threadIdx.x & 31;
  const int m    = lane & 15;
  const int koff = (lane & 16) ? 8 : 0;
  const __bf16* p = src + (size_t)m * lda + koff;
  v8bf lo = *(const v8bf*)(p);
  v8bf hi = *(const v8bf*)(p + 16);
  v16bf r;
#pragma unroll
  for (int i = 0; i < 8; ++i) { r[i] = lo[i]; r[i + 8] = hi[i]; }
  return r;
}

// B[kk][n] = srcT[n*ldb + kk]; lane n (0-15): K 0-7,16-23, lane 16+n: K 8-15,24-31.
__device__ inline v16bf load_b32x16(const __bf16* srcT, int ldb) {
  const int lane = threadIdx.x & 31;
  const int n    = lane & 15;
  const int koff = (lane & 16) ? 8 : 0;
  const __bf16* p = srcT + (size_t)n * ldb + koff;
  v8bf lo = *(const v8bf*)(p);
  v8bf hi = *(const v8bf*)(p + 16);
  v16bf r;
#pragma unroll
  for (int i = 0; i < 8; ++i) { r[i] = lo[i]; r[i + 8] = hi[i]; }
  return r;
}

// ---------------------------------------------------------------------------
__global__ void cvt_f32_bf16(const float* __restrict__ s, __bf16* __restrict__ d, int n) {
  int i = blockIdx.x * blockDim.x + threadIdx.x;
  if (i < n) d[i] = (__bf16)s[i];
}

__global__ void transpose_w(const float* __restrict__ W, __bf16* __restrict__ Wt) {
  __shared__ float tile[32][33];
  const int bx = blockIdx.x * 32, by = blockIdx.y * 32;
  const int tx = threadIdx.x, ty = threadIdx.y;
#pragma unroll
  for (int i = 0; i < 32; i += 8)
    tile[ty + i][tx] = W[(size_t)(by + ty + i) * D + bx + tx];
  __syncthreads();
#pragma unroll
  for (int i = 0; i < 32; i += 8)
    Wt[(size_t)(bx + ty + i) * D + by + tx] = (__bf16)tile[tx][ty + i];
}

// ---------------------------------------------------------------------------
// y = A(bf16 MxK) @ Wt^T + bias.  Wt bf16 [N][K] row-major.
// block = 128 (4 waves); block tile 128x64; wave tile 32x64 (8 accumulators).
// B tile (64x32 = 4KB) async-staged to LDS, double-buffered, shared by all
// waves.  Branchless steady state: the "next" tile index wraps, so every
// iteration issues + WAIT_ASYNC(2); the one redundant tail fetch lands in the
// inactive buffer and drains under s_endpgm's implicit wait-idle.
// mode 0: Q [B,H,S,Dh]  1: K [B,H,S,Dh]  2: V^T [B,H,Dh,S]  3: fp32 [M][N]
__global__ __launch_bounds__(128, 1)
void gemm_bf16(const __bf16* __restrict__ A, const __bf16* __restrict__ Bt,
               const float* __restrict__ bias, void* __restrict__ outp,
               int mode) {
  __shared__ __align__(16) __bf16 btile[2][64 * 32];   // 4KB per buffer
  const int tid  = threadIdx.x;
  const int lane = tid & 31;
  const int wave = tid >> 5;
  const int r0 = blockIdx.y * 128 + wave * 32;
  const int c0 = blockIdx.x * 64;

  // per-thread async-copy addressing, hoisted out of the loop
  const int iA = tid, iB = tid + 128;
  const int rowA = iA >> 2, segA = iA & 3;       // 64 rows x 4 segs x 16B
  const int rowB = iB >> 2, segB = iB & 3;
  const char* gA = (const char*)Bt + ((size_t)(c0 + rowA) * D + segA * 8) * 2;
  const char* gB = (const char*)Bt + ((size_t)(c0 + rowB) * D + segB * 8) * 2;
  const unsigned lA = (unsigned)(uintptr_t)&btile[0][0] + (unsigned)(rowA * 64 + segA * 16);
  const unsigned lB = (unsigned)(uintptr_t)&btile[0][0] + (unsigned)(rowB * 64 + segB * 16);

  auto issue = [&](int buf, int k) {
    const unsigned lo = (unsigned)(buf << 12);
    async_b128(lA + lo, gA + (size_t)k * 2);
    async_b128(lB + lo, gB + (size_t)k * 2);
  };

  issue(0, 0);
  v8f acc[2][4] = {};
  const __bf16* ap = A + (size_t)r0 * D;

#pragma unroll 1
  for (int ks = 0; ks < D / 32; ++ks) {
    const int k   = ks * 32;
    const int buf = ks & 1;
    issue(buf ^ 1, (k + 32) & (D - 1));   // wraps on last iter: harmless refetch
    WAIT_ASYNC(2);
    __syncthreads();

    __builtin_prefetch(ap + k + 64, 0, 1);
    v16bf a0 = load_a16x32(ap + k, D);
    v16bf a1 = load_a16x32(ap + 16 * D + k, D);
    const __bf16* bt = &btile[buf][0];
#pragma unroll
    for (int t = 0; t < 4; ++t) {
      v16bf b = load_b32x16(bt + 16 * t * 32, 32);
      acc[0][t] = WMMA_BF16(a0, b, acc[0][t]);
      acc[1][t] = WMMA_BF16(a1, b, acc[1][t]);
    }
    __syncthreads();
  }

  const int n  = lane & 15;
  const int mb = (lane & 16) ? 8 : 0;
#pragma unroll
  for (int u = 0; u < 2; ++u)
#pragma unroll
    for (int t = 0; t < 4; ++t) {
      const int col = c0 + 16 * t + n;
      const float bv = bias[col];
#pragma unroll
      for (int r = 0; r < 8; ++r) {
        const int row = r0 + 16 * u + mb + r;
        const float v = acc[u][t][r] + bv;
        if (mode == 3) {
          ((float*)outp)[(size_t)row * D + col] = v;
        } else {
          const int b = row >> 11, s = row & (S - 1);
          const int h = col >> 6,  d = col & (Dh - 1);
          __bf16* o = (__bf16*)outp;
          if (mode == 2)
            o[((size_t)(b * H + h) * Dh + d) * S + s] = (__bf16)v;   // V^T
          else
            o[((size_t)(b * H + h) * S + s) * Dh + d] = (__bf16)v;   // Q / K
        }
      }
    }
}

// ---------------------------------------------------------------------------
// Flash attention.  grid = B*H*(S/64), block = 128 (4 waves x 16 q-rows).
// Per 32-key step: K tile (32x64, contiguous 4KB) and V^T tile (64x32,
// strided) async-staged to LDS (double-buffered, shared by all waves);
// 4 WMMAs scores + online softmax + P->LDS relayout + 4 WMMAs P@V.
__global__ __launch_bounds__(128, 1)
void flash_attn(const __bf16* __restrict__ Qb, const __bf16* __restrict__ Kb,
                const __bf16* __restrict__ Vt, __bf16* __restrict__ ctx) {
  __shared__ __align__(16) __bf16 ktile[2][32 * 64];   // [key][d]
  __shared__ __align__(16) __bf16 vtile[2][64 * 32];   // [d][key]
  __shared__ __align__(16) __bf16 pbuf[4][16 * 32];    // per-wave P staging
  const int tid  = threadIdx.x;
  const int lane = tid & 31;
  const int wave = tid >> 5;
  const int bh = blockIdx.x >> 5;
  const int qt = blockIdx.x & 31;
  const int q0 = qt * 64 + wave * 16;

  const __bf16* Qp = Qb + ((size_t)bh * S + q0) * Dh;
  const __bf16* Kp = Kb + (size_t)bh * S * Dh;
  const __bf16* Vp = Vt + (size_t)bh * Dh * S;

  // per-thread async-copy addressing, hoisted out of the loop
  const int i0 = tid, i1 = tid + 128;
  const int rv0 = i0 >> 2, sv0 = i0 & 3;
  const int rv1 = i1 >> 2, sv1 = i1 & 3;
  const char* kg0 = (const char*)Kp + (size_t)i0 * 16;
  const char* kg1 = (const char*)Kp + (size_t)i1 * 16;
  const char* vg0 = (const char*)Vp + (size_t)rv0 * (S * 2) + sv0 * 16;
  const char* vg1 = (const char*)Vp + (size_t)rv1 * (S * 2) + sv1 * 16;
  const unsigned kl0 = (unsigned)(uintptr_t)&ktile[0][0] + (unsigned)(i0 * 16);
  const unsigned kl1 = (unsigned)(uintptr_t)&ktile[0][0] + (unsigned)(i1 * 16);
  const unsigned vl0 = (unsigned)(uintptr_t)&vtile[0][0] + (unsigned)(rv0 * 64 + sv0 * 16);
  const unsigned vl1 = (unsigned)(uintptr_t)&vtile[0][0] + (unsigned)(rv1 * 64 + sv1 * 16);

  auto issue = [&](int buf, int kt) {
    const unsigned lo = (unsigned)(buf << 12);
    const size_t ko = (size_t)kt * (Dh * 2);   // K tile: contiguous, 128B/key
    const size_t vo = (size_t)kt * 2;          // V^T tile: +kt columns
    async_b128(kl0 + lo, kg0 + ko);
    async_b128(kl1 + lo, kg1 + ko);
    async_b128(vl0 + lo, vg0 + vo);
    async_b128(vl1 + lo, vg1 + vo);
  };

  issue(0, 0);
  const v16bf qa0 = load_a16x32(Qp, Dh);
  const v16bf qa1 = load_a16x32(Qp + 32, Dh);

  v8f acc[4] = {};
  float m_i[8], l_i[8];
#pragma unroll
  for (int r = 0; r < 8; ++r) { m_i[r] = -1e30f; l_i[r] = 0.0f; }
  const float scale = 0.125f;  // 1/sqrt(Dh)

  const int n  = lane & 15;
  const int mb = (lane & 16) ? 8 : 0;
  __bf16* pw = pbuf[wave];

#pragma unroll 1
  for (int ks = 0; ks < S / 32; ++ks) {
    const int buf = ks & 1;
    issue(buf ^ 1, (ks * 32 + 32) & (S - 1));   // wraps on last iter
    WAIT_ASYNC(4);
    __syncthreads();

    const __bf16* kl = &ktile[buf][0];
    const __bf16* vl = &vtile[buf][0];
    // ---- scores ---------------------------------------------------------
    v8f c0 = {}, c1 = {};
    {
      v16bf bk = load_b32x16(kl, 64);                 // keys 0..15, d 0..31
      c0 = WMMA_BF16(qa0, bk, c0);
      bk = load_b32x16(kl + 32, 64);                  // d 32..63
      c0 = WMMA_BF16(qa1, bk, c0);
      bk = load_b32x16(kl + 16 * 64, 64);             // keys 16..31
      c1 = WMMA_BF16(qa0, bk, c1);
      bk = load_b32x16(kl + 16 * 64 + 32, 64);
      c1 = WMMA_BF16(qa1, bk, c1);
    }
    // ---- online softmax -------------------------------------------------
#pragma unroll
    for (int r = 0; r < 8; ++r) {
      const float s0 = c0[r] * scale, s1 = c1[r] * scale;
      float rm = fmaxf(s0, s1);
      rm = fmaxf(rm, __shfl_xor(rm, 1));
      rm = fmaxf(rm, __shfl_xor(rm, 2));
      rm = fmaxf(rm, __shfl_xor(rm, 4));
      rm = fmaxf(rm, __shfl_xor(rm, 8));
      const float mn = fmaxf(m_i[r], rm);
      const float alpha = __expf(m_i[r] - mn);
      const float p0 = __expf(s0 - mn), p1 = __expf(s1 - mn);
      float rs = p0 + p1;
      rs += __shfl_xor(rs, 1);
      rs += __shfl_xor(rs, 2);
      rs += __shfl_xor(rs, 4);
      rs += __shfl_xor(rs, 8);
      l_i[r] = l_i[r] * alpha + rs;
      m_i[r] = mn;
      acc[0][r] *= alpha; acc[1][r] *= alpha;
      acc[2][r] *= alpha; acc[3][r] *= alpha;
      pw[(mb + r) * 32 + n]      = (__bf16)p0;
      pw[(mb + r) * 32 + 16 + n] = (__bf16)p1;
    }
    // ---- O += P @ V -----------------------------------------------------
    const v16bf pa = load_a16x32(pw, 32);   // same-wave LDS RAW, DScnt-ordered
#pragma unroll
    for (int t = 0; t < 4; ++t) {
      v16bf bv = load_b32x16(vl + 16 * t * 32, 32);
      acc[t] = WMMA_BF16(pa, bv, acc[t]);
    }
    __syncthreads();
  }

  const int b = bh >> 4, h = bh & (H - 1);
#pragma unroll
  for (int t = 0; t < 4; ++t)
#pragma unroll
    for (int r = 0; r < 8; ++r) {
      const float v = acc[t][r] / l_i[r];
      const int row = q0 + mb + r;
      ctx[((size_t)(b * S + row)) * D + h * Dh + 16 * t + n] = (__bf16)v;
    }
}

// ---------------------------------------------------------------------------
extern "C" void kernel_launch(void* const* d_in, const int* in_sizes, int n_in,
                              void* d_out, int out_size, void* d_ws, size_t ws_size,
                              hipStream_t stream) {
  (void)in_sizes; (void)n_in; (void)out_size;
  if (ws_size < WS_NEED) return;

  const float* x  = (const float*)d_in[0];
  const float* wq = (const float*)d_in[1];
  const float* bq = (const float*)d_in[2];
  const float* wk = (const float*)d_in[3];
  const float* bk = (const float*)d_in[4];
  const float* wv = (const float*)d_in[5];
  const float* bv = (const float*)d_in[6];
  const float* wo = (const float*)d_in[7];
  const float* bo = (const float*)d_in[8];

  char* ws = (char*)d_ws;
  __bf16* xb   = (__bf16*)(ws + OFF_XB);
  __bf16* wtq  = (__bf16*)(ws + OFF_WTQ);
  __bf16* wtk  = (__bf16*)(ws + OFF_WTK);
  __bf16* wtv  = (__bf16*)(ws + OFF_WTV);
  __bf16* wto  = (__bf16*)(ws + OFF_WTO);
  __bf16* qb   = (__bf16*)(ws + OFF_QB);
  __bf16* kb   = (__bf16*)(ws + OFF_KB);
  __bf16* vt   = (__bf16*)(ws + OFF_VT);
  __bf16* ctxb = (__bf16*)(ws + OFF_CTX);

  cvt_f32_bf16<<<(M * D) / 256, 256, 0, stream>>>(x, xb, M * D);

  dim3 tb(32, 8), tg(D / 32, D / 32);
  transpose_w<<<tg, tb, 0, stream>>>(wq, wtq);
  transpose_w<<<tg, tb, 0, stream>>>(wk, wtk);
  transpose_w<<<tg, tb, 0, stream>>>(wv, wtv);
  transpose_w<<<tg, tb, 0, stream>>>(wo, wto);

  dim3 gg(D / 64, M / 128);
  gemm_bf16<<<gg, 128, 0, stream>>>(xb, wtq, bq, qb, 0);
  gemm_bf16<<<gg, 128, 0, stream>>>(xb, wtk, bk, kb, 1);
  gemm_bf16<<<gg, 128, 0, stream>>>(xb, wtv, bv, vt, 2);

  flash_attn<<<Bn * H * (S / 64), 128, 0, stream>>>(qb, kb, vt, ctxb);

  gemm_bf16<<<gg, 128, 0, stream>>>(ctxb, wto, bo, d_out, 3);
}
